// TSGNet_231928234727
// MI455X (gfx1250) — matrix operand order repaired and checked by reading, not compiled, not run on hardware
//
#include <hip/hip_runtime.h>
#include <hip/hip_bf16.h>
#include <math.h>

// ---------------------------------------------------------------------------
// TSGNet on gfx1250 (MI455X): GCN x8 + static MLP + LSTM + final projection.
// All GEMMs via V_WMMA_F32_16X16X32_BF16 (wave32), fp32->bf16 staged in LDS,
// software-pipelined: next K-tile's global loads issue before current WMMAs.
// ---------------------------------------------------------------------------

#define T_     8
#define N_     10000
#define F_IN_  128
#define H_     128
#define HID_   1024
#define HID4_  256
#define C_     16
#define E_     320000

typedef __attribute__((ext_vector_type(16))) __bf16 v16bf;
typedef __attribute__((ext_vector_type(8)))  __bf16 v8bf;
typedef __attribute__((ext_vector_type(8)))  float  v8f;

// GEMM flags
#define GF_B_TRANS 1   // B stored [Nc,K] row-major; compute A @ B^T
#define GF_BIAS    2
#define GF_RELU    4
#define GF_ACCUM   8   // C += A@B (read existing C)

#define BM  64
#define BN  128
#define BKK 64          // two WMMA k-steps (32 each) per LDS stage

// branchless guarded load: clamped address + cndmask zeroing (no exec branches)
__device__ __forceinline__ float ld_guard(const float* __restrict__ p,
                                          int r, int c, int R, int Cc) {
    int rr = r < R ? r : R - 1;
    int cc = c < Cc ? c : Cc - 1;
    float v = p[(size_t)rr * Cc + cc];
    return (r < R && c < Cc) ? v : 0.0f;
}

__device__ __forceinline__ v8bf pack8(const float4 a, const float4 b) {
    v8bf o;
    o[0] = (__bf16)a.x; o[1] = (__bf16)a.y; o[2] = (__bf16)a.z; o[3] = (__bf16)a.w;
    o[4] = (__bf16)b.x; o[5] = (__bf16)b.y; o[6] = (__bf16)b.z; o[7] = (__bf16)b.w;
    return o;
}

// ---------------------------------------------------------------------------
// Tiled bf16 WMMA GEMM: C[M,Nc] = A[M,K] @ B (+bias)(+relu)(+=)
// Block: 64x128 tile, 256 threads = 8 wave32 waves.
// Wave w: m-frag mr=w&3, n-frags nb..nb+3 (nb=(w>>2)*4) -> 8 WMMA/stage.
// LDS tiles in per-lane fragment order (contiguous k = 16*(lane/16) + h): each lane's v16bf fragment
// is one contiguous 32B chunk (2x ds_load_b128).
// Grid: x = N tiles (fast) so concurrent blocks share the same A strip in L2.
// Pipeline: tile k+1 global_load_b128 -> regs issued before tile-k WMMAs.
// ---------------------------------------------------------------------------
__global__ __launch_bounds__(256) void gemm_bf16_wmma(
    const float* __restrict__ A, const float* __restrict__ B,
    const float* __restrict__ bias, float* __restrict__ C,
    int M, int Nc, int K, int flags)
{
    __shared__ __bf16 sA[2 * 4 * 32 * 16];   // [ksub][mfrag][lane][h] = 8KB
    __shared__ __bf16 sB[2 * 8 * 32 * 16];   // [ksub][nfrag][lane][h] = 16KB

    const int tid  = threadIdx.x;
    const int lane = tid & 31;
    const int wave = tid >> 5;
    const int bn = blockIdx.x * BN;
    const int bm = blockIdx.y * BM;

    const int mr = wave & 3;
    const int nb = (wave >> 2) * 4;

    v8f acc[4] = {{}, {}, {}, {}};

    // ---- A staging coords: each thread owns ONE fragment row (16 halves)
    // sA[((ks*4+f)*32+lane)*16 + h]; halves 0..15 <-> k=kb..kb+15,
    // kb = k0 + ks*32 + (lane&16 ? 16 : 0)
    const int a_lane = tid & 31;
    const int a_f    = (tid >> 5) & 3;
    const int a_ks   = tid >> 7;                    // 0..1
    const int a_m    = bm + a_f * 16 + (a_lane & 15);
    const int a_koff = a_ks * 32 + ((a_lane & 16) ? 16 : 0);
    __bf16* a_dst = &sA[((a_ks * 4 + a_f) * 32 + a_lane) * 16];

    // ---- B staging coords: each thread owns TWO fragment rows
    // sB[((ks*8+f)*32+lane)*16 + h]; halves 0..15 <-> k = kb..kb+15,
    // kb = k0 + ks*32 + (lane&16 ? 16 : 0)
    int b_ln[2], b_n[2], b_koff[2];
    __bf16* b_dst[2];
    #pragma unroll
    for (int rr = 0; rr < 2; ++rr) {
        const int r = tid * 2 + rr;
        const int ln = r & 31;
        const int f  = (r >> 5) & 7;
        const int ks = r >> 8;
        b_ln[rr]   = ln;
        b_n[rr]    = bn + f * 16 + (ln & 15);
        b_koff[rr] = ks * 32 + ((ln & 16) ? 16 : 0);
        b_dst[rr]  = &sB[((ks * 8 + f) * 32 + ln) * 16];
    }

    const bool fullM = (bm + BM) <= M;
    const bool fullN = (bn + BN) <= Nc;
    const bool btrans = (flags & GF_B_TRANS) != 0;

    // register double-buffer for the next tile (raw fp32; converted at store)
    float4 ra[4];
    float4 rb[2][4];

    // ---- tile loader: global -> registers (issues loads, no LDS traffic)
    auto load_tile = [&](int k0) {
        const bool fullK = (k0 + BKK) <= K;
        if (fullM && fullK) {
            const float* pa = A + (size_t)a_m * K + k0 + a_koff;
            ra[0] = *(const float4*)(pa + 0);
            ra[1] = *(const float4*)(pa + 4);
            ra[2] = *(const float4*)(pa + 8);
            ra[3] = *(const float4*)(pa + 12);
        } else {
            float t[16];
            #pragma unroll
            for (int h = 0; h < 8; ++h)
                t[h] = ld_guard(A, a_m, k0 + a_koff + h, M, K);
            #pragma unroll
            for (int h = 0; h < 8; ++h)
                t[8 + h] = ld_guard(A, a_m, k0 + a_koff + 8 + h, M, K);
            ra[0] = make_float4(t[0], t[1], t[2], t[3]);
            ra[1] = make_float4(t[4], t[5], t[6], t[7]);
            ra[2] = make_float4(t[8], t[9], t[10], t[11]);
            ra[3] = make_float4(t[12], t[13], t[14], t[15]);
        }
        #pragma unroll
        for (int rr = 0; rr < 2; ++rr) {
            const int n  = b_n[rr];
            const int kb = k0 + b_koff[rr];
            if (btrans && fullN && fullK) {
                const float* pb = B + (size_t)n * K + kb;
                rb[rr][0] = *(const float4*)(pb + 0);
                rb[rr][1] = *(const float4*)(pb + 4);
                rb[rr][2] = *(const float4*)(pb + 8);
                rb[rr][3] = *(const float4*)(pb + 12);
            } else {
                float t[16];
                #pragma unroll
                for (int h = 0; h < 16; ++h)
                    t[h] = btrans ? ld_guard(B, n, kb + h, Nc, K)
                                  : ld_guard(B, kb + h, n, K, Nc);
                rb[rr][0] = make_float4(t[0], t[1], t[2], t[3]);
                rb[rr][1] = make_float4(t[4], t[5], t[6], t[7]);
                rb[rr][2] = make_float4(t[8], t[9], t[10], t[11]);
                rb[rr][3] = make_float4(t[12], t[13], t[14], t[15]);
            }
        }
    };

    // ---- tile committer: registers -> bf16 -> LDS (fragment layout)
    auto store_tile = [&]() {
        *(v8bf*)(a_dst + 0) = pack8(ra[0], ra[1]);
        *(v8bf*)(a_dst + 8) = pack8(ra[2], ra[3]);
        #pragma unroll
        for (int rr = 0; rr < 2; ++rr) {
            *(v8bf*)(b_dst[rr] + 0) = pack8(rb[rr][0], rb[rr][1]);
            *(v8bf*)(b_dst[rr] + 8) = pack8(rb[rr][2], rb[rr][3]);
        }
    };

    load_tile(0);                                   // prologue
    for (int k0 = 0; k0 < K; k0 += BKK) {
        store_tile();
        __syncthreads();

        if (k0 + BKK < K)                           // issue next tile's loads
            load_tile(k0 + BKK);                    // (waited at next store)

        #pragma unroll
        for (int ks = 0; ks < 2; ++ks) {
            v16bf a = *(const v16bf*)&sA[((ks * 4 + mr) * 32 + lane) * 16];
            #pragma unroll
            for (int q = 0; q < 4; ++q) {
                v16bf b = *(const v16bf*)&sB[((ks * 8 + nb + q) * 32 + lane) * 16];
                acc[q] = __builtin_amdgcn_wmma_f32_16x16x32_bf16(
                             false, a, false, b, (short)0, acc[q], false, false);
            }
        }
        __syncthreads();
    }

    // ---- writeback (C frag: VGPR r -> M = r + (lane&16?8:0); N = lane&15)
    const int mbase = bm + mr * 16 + ((lane & 16) ? 8 : 0);
    const int nloc  = lane & 15;
    #pragma unroll
    for (int q = 0; q < 4; ++q) {
        const int n = bn + (nb + q) * 16 + nloc;
        if (n >= Nc) continue;
        #pragma unroll
        for (int r = 0; r < 8; ++r) {
            int m = mbase + r;
            if (m >= M) continue;
            float v = acc[q][r];
            if (flags & GF_ACCUM) v += C[(size_t)m * Nc + n];
            if (flags & GF_BIAS)  v += bias[n];
            if (flags & GF_RELU)  v = fmaxf(v, 0.0f);
            C[(size_t)m * Nc + n] = v;
        }
    }
}

// ---------------------------------------------------------------------------
// Pointwise / graph kernels
// ---------------------------------------------------------------------------
__global__ void fill_kernel(float* __restrict__ p, float val, int n) {
    int i = blockIdx.x * blockDim.x + threadIdx.x;
    if (i < n) p[i] = val;
}

__global__ void copy_kernel(float* __restrict__ d, const float* __restrict__ s, int n) {
    int i = blockIdx.x * blockDim.x + threadIdx.x;
    if (i < n) d[i] = s[i];
}

__global__ void deg_kernel(const int* __restrict__ dst, float* __restrict__ deg, int e) {
    int i = blockIdx.x * blockDim.x + threadIdx.x;
    if (i < e) atomicAdd(&deg[dst[i]], 1.0f);
}

__global__ void dinv_kernel(float* __restrict__ deg, int n) {
    int i = blockIdx.x * blockDim.x + threadIdx.x;
    if (i < n) {
        float d = deg[i];
        deg[i] = (d > 0.0f) ? rsqrtf(d) : 0.0f;
    }
}

// out[dst,:] += XW[src,:] * dinv[src]*dinv[dst]   (H=128 channels per edge)
__global__ void scatter_kernel(const float* __restrict__ xw,
                               const int* __restrict__ src,
                               const int* __restrict__ dst,
                               const float* __restrict__ dinv,
                               float* __restrict__ outp, int e) {
    int tid = blockIdx.x * blockDim.x + threadIdx.x;
    int edge = tid >> 7;
    int ch   = tid & 127;
    if (edge >= e) return;
    int s = src[edge], d = dst[edge];
    float norm = dinv[s] * dinv[d];
    atomicAdd(&outp[(size_t)d * H_ + ch], xw[(size_t)s * H_ + ch] * norm);
}

// self loop: out[i,:] += XW[i,:]*dinv[i]^2, optional relu (applied to total)
__global__ void selfloop_kernel(const float* __restrict__ xw,
                                const float* __restrict__ dinv,
                                float* __restrict__ outp, int n, int do_relu) {
    int tid = blockIdx.x * blockDim.x + threadIdx.x;
    int row = tid >> 7;
    int ch  = tid & 127;
    if (row >= n) return;
    float di = dinv[row];
    float v = outp[(size_t)row * H_ + ch] + xw[(size_t)row * H_ + ch] * di * di;
    if (do_relu) v = fmaxf(v, 0.0f);
    outp[(size_t)row * H_ + ch] = v;
}

__device__ __forceinline__ float sigm(float x) { return 1.0f / (1.0f + __expf(-x)); }

// torch gate order i,f,g,o ; gates [N,512]
__global__ void lstm_cell_kernel(const float* __restrict__ gates,
                                 float* __restrict__ h, float* __restrict__ c, int n) {
    int tid = blockIdx.x * blockDim.x + threadIdx.x;
    int row = tid >> 7;
    int col = tid & 127;
    if (row >= n) return;
    const float* g = gates + (size_t)row * (4 * H_);
    float ig = g[col];
    float fg = g[H_ + col];
    float gg = g[2 * H_ + col];
    float og = g[3 * H_ + col];
    size_t idx = (size_t)row * H_ + col;
    float cc = sigm(fg) * c[idx] + sigm(ig) * tanhf(gg);
    float hh = sigm(og) * tanhf(cc);
    c[idx] = cc;
    h[idx] = hh;
}

// log_softmax over 128-wide rows, one wave32 per row; optional additive input
__global__ __launch_bounds__(256) void log_softmax128_kernel(
    const float* __restrict__ in, const float* __restrict__ addv,
    float* __restrict__ outp, int n) {
    int row  = (blockIdx.x * blockDim.x + threadIdx.x) >> 5;
    int lane = threadIdx.x & 31;
    if (row >= n) return;
    const float* r = in + (size_t)row * H_;
    float v[4];
    float mx = -3.4e38f;
    #pragma unroll
    for (int j = 0; j < 4; ++j) { v[j] = r[lane + 32 * j]; mx = fmaxf(mx, v[j]); }
    #pragma unroll
    for (int o = 16; o > 0; o >>= 1) mx = fmaxf(mx, __shfl_xor(mx, o, 32));
    float s = 0.0f;
    #pragma unroll
    for (int j = 0; j < 4; ++j) s += __expf(v[j] - mx);
    #pragma unroll
    for (int o = 16; o > 0; o >>= 1) s += __shfl_xor(s, o, 32);
    float lse = mx + __logf(s);
    float* w = outp + (size_t)row * H_;
    #pragma unroll
    for (int j = 0; j < 4; ++j) {
        float o = v[j] - lse;
        if (addv) o += addv[(size_t)row * H_ + lane + 32 * j];
        w[lane + 32 * j] = o;
    }
}

// log_softmax over C=16 rows, one thread per row
__global__ void log_softmax16_kernel(const float* __restrict__ in,
                                     float* __restrict__ outp, int n) {
    int i = blockIdx.x * blockDim.x + threadIdx.x;
    if (i >= n) return;
    const float* r = in + (size_t)i * C_;
    float mx = -3.4e38f;
    #pragma unroll
    for (int j = 0; j < C_; ++j) mx = fmaxf(mx, r[j]);
    float s = 0.0f;
    #pragma unroll
    for (int j = 0; j < C_; ++j) s += __expf(r[j] - mx);
    float lse = mx + __logf(s);
    #pragma unroll
    for (int j = 0; j < C_; ++j) outp[(size_t)i * C_ + j] = r[j] - lse;
}

// ---------------------------------------------------------------------------
// Host orchestration
// ---------------------------------------------------------------------------
static inline dim3 gemm_grid(int M, int Nc) {
    return dim3((Nc + BN - 1) / BN, (M + BM - 1) / BM);
}

extern "C" void kernel_launch(void* const* d_in, const int* in_sizes, int n_in,
                              void* d_out, int out_size, void* d_ws, size_t ws_size,
                              hipStream_t stream) {
    (void)in_sizes; (void)n_in; (void)out_size; (void)ws_size;

    const float* t_x  = (const float*)d_in[0];
    const int*   t_ei = (const int*)  d_in[1];
    const float* s_x  = (const float*)d_in[2];
    const float* h0   = (const float*)d_in[3];
    const float* c0   = (const float*)d_in[4];
    const float* gW0  = (const float*)d_in[5];
    const float* gW1  = (const float*)d_in[6];
    const float* Wih  = (const float*)d_in[7];
    const float* Whh  = (const float*)d_in[8];
    const float* bih  = (const float*)d_in[9];
    const float* bhh  = (const float*)d_in[10];
    const float* nW0  = (const float*)d_in[11];
    const float* nb0  = (const float*)d_in[12];
    const float* nW1  = (const float*)d_in[13];
    const float* nb1  = (const float*)d_in[14];
    const float* nW2  = (const float*)d_in[15];
    const float* nb2  = (const float*)d_in[16];
    const float* finW = (const float*)d_in[17];
    const float* finb = (const float*)d_in[18];
    float* out = (float*)d_out;
    float* ws  = (float*)d_ws;

    // workspace layout (floats); all offsets 16B-aligned
    float* t_all = ws;                                  // [T,N,H]
    float* ws1   = t_all + (size_t)T_ * N_ * H_;        // [N,HID] multi-use
    float* ws2   = ws1 + (size_t)N_ * HID_;             // [N,HID/4] multi-use
    float* xstat = ws2 + (size_t)N_ * HID4_;            // [N,H]
    float* hbuf  = xstat + (size_t)N_ * H_;             // [N,H]
    float* cbuf  = hbuf + (size_t)N_ * H_;              // [N,H]

    float* XW  = ws1;                     // [N,H] GCN layer output pre-agg
    float* H1  = ws1 + (size_t)N_ * H_;   // [N,H] GCN aggregated
    float* deg = ws2;                     // [N]

    const int TPB = 256;
    const int nNH   = N_ * H_;
    const int nEdge = (E_ + TPB - 1) / TPB;
    const int nScat = (E_ * 128 + TPB - 1) / TPB;
    const int nNH_b = (nNH + TPB - 1) / TPB;
    const int nN_b  = (N_ + TPB - 1) / TPB;

    // ================= GCN temporal encoder (T windows) =================
    for (int t = 0; t < T_; ++t) {
        const float* xt  = t_x + (size_t)t * N_ * F_IN_;
        const float* W0t = gW0 + (size_t)t * F_IN_ * H_;   // [K,Nc]
        const float* W1t = gW1 + (size_t)t * H_ * H_;      // [K,Nc]
        const int* src = t_ei + (size_t)t * 2 * E_;
        const int* dst = src + E_;
        float* tout = t_all + (size_t)t * N_ * H_;

        // XW = x @ W0
        gemm_bf16_wmma<<<gemm_grid(N_, H_), TPB, 0, stream>>>(
            xt, W0t, nullptr, XW, N_, H_, F_IN_, 0);
        // degree (self loop contributes 1) -> dinv
        fill_kernel<<<nN_b, TPB, 0, stream>>>(deg, 1.0f, N_);
        deg_kernel<<<nEdge, TPB, 0, stream>>>(dst, deg, E_);
        dinv_kernel<<<nN_b, TPB, 0, stream>>>(deg, N_);
        // layer 0 aggregation + relu
        fill_kernel<<<nNH_b, TPB, 0, stream>>>(H1, 0.0f, nNH);
        scatter_kernel<<<nScat, TPB, 0, stream>>>(XW, src, dst, deg, H1, E_);
        selfloop_kernel<<<nNH_b, TPB, 0, stream>>>(XW, deg, H1, N_, 1);
        // layer 1: XW2 = H1 @ W1 (reuse XW buffer)
        gemm_bf16_wmma<<<gemm_grid(N_, H_), TPB, 0, stream>>>(
            H1, W1t, nullptr, XW, N_, H_, H_, 0);
        fill_kernel<<<nNH_b, TPB, 0, stream>>>(tout, 0.0f, nNH);
        scatter_kernel<<<nScat, TPB, 0, stream>>>(XW, src, dst, deg, tout, E_);
        selfloop_kernel<<<nNH_b, TPB, 0, stream>>>(XW, deg, tout, N_, 0);
    }

    // ================= static encoder MLP (dominant GEMM) =================
    float* A1 = ws1;   // [N,1024]
    float* A2 = ws2;   // [N,256]
    gemm_bf16_wmma<<<gemm_grid(N_, HID_), TPB, 0, stream>>>(
        s_x, nW0, nb0, A1, N_, HID_, N_, GF_B_TRANS | GF_BIAS | GF_RELU);
    gemm_bf16_wmma<<<gemm_grid(N_, HID4_), TPB, 0, stream>>>(
        A1, nW1, nb1, A2, N_, HID4_, HID_, GF_B_TRANS | GF_BIAS);
    gemm_bf16_wmma<<<gemm_grid(N_, H_), TPB, 0, stream>>>(
        A2, nW2, nb2, xstat, N_, H_, HID4_, GF_B_TRANS | GF_BIAS);
    // x = log_softmax(x) in place (each wave reads its row before writing)
    log_softmax128_kernel<<<(N_ * 32 + TPB - 1) / TPB, TPB, 0, stream>>>(
        xstat, nullptr, xstat, N_);

    // ================= LSTM over T steps =================
    float* gates = ws1;  // [N,512]
    copy_kernel<<<nNH_b, TPB, 0, stream>>>(hbuf, h0, nNH);
    copy_kernel<<<nNH_b, TPB, 0, stream>>>(cbuf, c0, nNH);
    for (int t = 0; t < T_; ++t) {
        const float* xt = t_all + (size_t)t * N_ * H_;
        gemm_bf16_wmma<<<gemm_grid(N_, 4 * H_), TPB, 0, stream>>>(
            xt, Wih, bih, gates, N_, 4 * H_, H_, GF_B_TRANS | GF_BIAS);
        gemm_bf16_wmma<<<gemm_grid(N_, 4 * H_), TPB, 0, stream>>>(
            hbuf, Whh, bhh, gates, N_, 4 * H_, H_, GF_B_TRANS | GF_BIAS | GF_ACCUM);
        lstm_cell_kernel<<<nNH_b, TPB, 0, stream>>>(gates, hbuf, cbuf, N_);
    }

    // z = log_softmax(hT); zx = z + xstat
    float* zx = ws1;  // [N,H]
    log_softmax128_kernel<<<(N_ * 32 + TPB - 1) / TPB, TPB, 0, stream>>>(
        hbuf, xstat, zx, N_);

    // out = log_softmax(zx @ finW^T + finb)
    float* o16 = ws2;  // [N,16]
    gemm_bf16_wmma<<<gemm_grid(N_, C_), TPB, 0, stream>>>(
        zx, finW, finb, o16, N_, C_, H_, GF_B_TRANS | GF_BIAS);
    log_softmax16_kernel<<<nN_b, TPB, 0, stream>>>(o16, out, N_);
}
